// GRUBaseline_62603443306683
// MI455X (gfx1250) — compile-verified, matrix operand order checked by reading, not run on hardware
//
#include <hip/hip_runtime.h>
#include <hip/hip_bf16.h>

// ---------------------------------------------------------------------------
// Fused 2-layer GRU + head for MI455X (gfx1250, wave32, WMMA).
//
// Design:
//  * One kernel runs both GRU layers in lockstep over t (out0 never hits HBM).
//  * Recurrent GEMMs use v_wmma_f32_16x16x32_f16 (f16 in, f32 accum).
//  * Hidden states h0/h1 live in LDS as f16; A-fragments built with aligned
//    ds_load_b32 gathers matching the 16-bit A-matrix VGPR layout.
//  * Weight B-fragments pre-packed (prep kernel) into the 32x16 WMMA lane
//    layout and kept register-resident for the whole time loop.
//  * Block = 16 batch rows x 4 waves; wave nj owns 16 columns of each gate,
//    so r/z/n columns align per-lane for the gate nonlinearity.
// ---------------------------------------------------------------------------

#define B_  2048
#define L_  256
#define C_  5
#define H_  64

typedef __attribute__((ext_vector_type(16))) _Float16 v16h;
typedef __attribute__((ext_vector_type(8)))  float    v8f;

// Workspace layout (in _Float16 units); all offsets 32-half (64B) aligned.
#define OFF_WHH0 0
#define OFF_WIH1 12288
#define OFF_WHH1 24576
#define OFF_W1   36864   // total 40960 halfs = 80 KB

// ---------------------------------------------------------------------------
// Prep: pack W (N x 64, f32) into WMMA B-fragment order [kb][ntile][lane][16]
// for B[k][n] = W[n][k].  16-bit B 32x16 layout (mirror of A layout):
//   lane = g*16 + n%16 ; half i: v=i/2,h=i%2 ; k = (v<4 ? 2v+h : 16+2(v-4)+h)+8g
// ---------------------------------------------------------------------------
__global__ void gru_pack_bfrag(const float* __restrict__ W,
                               _Float16* __restrict__ dst, int nTiles) {
  int idx = blockIdx.x * 256 + threadIdx.x;
  int total = 2 * nTiles * 32 * 16;
  if (idx >= total) return;
  int i    = idx & 15;
  int lane = (idx >> 4) & 31;
  int nt   = (idx >> 9) % nTiles;
  int kb   = idx / (512 * nTiles);
  int n    = nt * 16 + (lane & 15);
  int g    = lane >> 4;
  int v    = i >> 1, h = i & 1;
  int k    = kb * 32 + ((v < 4) ? (2 * v + h) : (16 + 2 * (v - 4) + h)) + 8 * g;
  dst[idx] = (_Float16)W[n * H_ + k];
}

// Build 16x32 f16 A-fragment from LDS h-buffer (row-major [16][64] f16).
// A layout: lane m=lane&15 (both halves), g=lane>>4; VGPR q holds K-pair
// (contiguous in memory -> aligned 32-bit LDS loads).
__device__ __forceinline__ v16h lds_a_frag(const _Float16* hb, int lane, int kb) {
  int m = lane & 15, g = (lane >> 4) & 1;
  union { v16h v; unsigned int u[8]; } r;
#pragma unroll
  for (int q = 0; q < 8; ++q) {
    int kp = ((q < 4) ? (2 * q) : (16 + 2 * (q - 4))) + 8 * g + 32 * kb;
    r.u[q] = *(const unsigned int*)(hb + m * H_ + kp);
  }
  return r.v;
}

__device__ __forceinline__ float fsig(float x) {
  return __builtin_amdgcn_rcpf(1.0f + __expf(-x));   // v_exp_f32 + v_rcp_f32
}
__device__ __forceinline__ float ftanh(float x) {
  return fmaf(2.0f, fsig(2.0f * x), -1.0f);
}

#define WMMA_F16(A, Bf, Cacc) \
  __builtin_amdgcn_wmma_f32_16x16x32_f16(false, (A), false, (Bf), (short)0, (Cacc), false, false)

__global__ void __launch_bounds__(128)
gru_fused(const float* __restrict__ x, const unsigned char* __restrict__ xmask,
          const float* __restrict__ wih0,
          const float* __restrict__ bih0, const float* __restrict__ bhh0,
          const float* __restrict__ bih1, const float* __restrict__ bhh1,
          const float* __restrict__ b1,  const float* __restrict__ w2,
          const float* __restrict__ b2,
          const _Float16* __restrict__ wsH, float* __restrict__ out) {
  __shared__ _Float16 h0b[16 * H_];   // layer-0 hidden tile (f16)
  __shared__ _Float16 h1b[16 * H_];   // layer-1 hidden tile (f16)
  __shared__ float    xb[16 * C_];    // x_t tile

  const int tid  = threadIdx.x;
  const int lane = tid & 31;
  const int nj   = tid >> 5;            // wave 0..3 -> owns cols [16nj,16nj+16)
  const int nl   = lane & 15;
  const int gg   = (lane >> 4) & 1;     // D-layout row half: m = r + 8*gg
  const int col  = nj * 16 + nl;        // column within H
  const int bbase = blockIdx.x * 16;

  // ---- register-resident weight B-fragments (18 x v16h = 144 VGPRs) ----
  v16h Whh0[2][3], Wih1[2][3], Whh1[2][3];
#pragma unroll
  for (int kb = 0; kb < 2; ++kb)
#pragma unroll
    for (int g3 = 0; g3 < 3; ++g3) {
      size_t o = ((size_t)(kb * 12 + (g3 * 4 + nj)) * 32 + lane) * 16;
      Whh0[kb][g3] = *(const v16h*)(wsH + OFF_WHH0 + o);
      Wih1[kb][g3] = *(const v16h*)(wsH + OFF_WIH1 + o);
      Whh1[kb][g3] = *(const v16h*)(wsH + OFF_WHH1 + o);
    }

  // per-lane loop-invariant scalars: input-projection row of w_ih0 + biases
  float w0c[3][C_], vbih0[3], vbhh0[3], vbih1[3], vbhh1[3];
#pragma unroll
  for (int g3 = 0; g3 < 3; ++g3) {
    int gc = g3 * H_ + col;
    vbih0[g3] = bih0[gc]; vbhh0[g3] = bhh0[gc];
    vbih1[g3] = bih1[gc]; vbhh1[g3] = bhh1[gc];
#pragma unroll
    for (int c = 0; c < C_; ++c) w0c[g3][c] = wih0[gc * C_ + c];
  }

  for (int i = tid; i < 16 * H_; i += 128) {
    h0b[i] = (_Float16)0.0f;
    h1b[i] = (_Float16)0.0f;
  }
  __syncthreads();

#pragma unroll 1
  for (int t = 0; t < L_; ++t) {
    // stage masked x_t tile (16 rows x 5 ch)
    if (tid < 16 * C_) {
      int row = tid / C_, c = tid - row * C_;
      size_t gi = ((size_t)(bbase + row) * L_ + t) * C_ + c;
      float xv = x[gi];
      xb[tid] = xmask[gi] ? xv : 0.0f;
    }
    __syncthreads();                       // bar1: x + prev h1 visible

    // ================= layer 0 =================
    v16h a0 = lds_a_frag(h0b, lane, 0);
    v16h a1 = lds_a_frag(h0b, lane, 1);
    v8f acc0[3];
#pragma unroll
    for (int g3 = 0; g3 < 3; ++g3) {
      v8f c8 = {};
      c8 = WMMA_F16(a0, Whh0[0][g3], c8);
      c8 = WMMA_F16(a1, Whh0[1][g3], c8);
      acc0[g3] = c8;
    }

    float hnew0[8];
#pragma unroll
    for (int r = 0; r < 8; ++r) {
      int m = r + 8 * gg;
      float xr = vbih0[0], xz = vbih0[1], xn = vbih0[2];
#pragma unroll
      for (int c = 0; c < C_; ++c) {        // K=5 input projection on VALU
        float xv = xb[m * C_ + c];
        xr = fmaf(xv, w0c[0][c], xr);
        xz = fmaf(xv, w0c[1][c], xz);
        xn = fmaf(xv, w0c[2][c], xn);
      }
      float rr = fsig(xr + acc0[0][r] + vbhh0[0]);
      float zz = fsig(xz + acc0[1][r] + vbhh0[1]);
      float nn = ftanh(xn + rr * (acc0[2][r] + vbhh0[2]));
      float hp = (float)h0b[m * H_ + col];
      hnew0[r] = (1.0f - zz) * nn + zz * hp;
    }
    __syncthreads();                       // bar2: all h0 reads done
#pragma unroll
    for (int r = 0; r < 8; ++r) h0b[(r + 8 * gg) * H_ + col] = (_Float16)hnew0[r];
    __syncthreads();                       // bar3: new h0 visible

    // ================= layer 1 =================
    v16h ax0 = lds_a_frag(h0b, lane, 0);   // out0_t = new h0
    v16h ax1 = lds_a_frag(h0b, lane, 1);
    v8f gx[3];
#pragma unroll
    for (int g3 = 0; g3 < 3; ++g3) {
      v8f c8 = {};
      c8 = WMMA_F16(ax0, Wih1[0][g3], c8);
      c8 = WMMA_F16(ax1, Wih1[1][g3], c8);
      gx[g3] = c8;
    }
    v16h ah0 = lds_a_frag(h1b, lane, 0);
    v16h ah1 = lds_a_frag(h1b, lane, 1);
    v8f gh[3];
#pragma unroll
    for (int g3 = 0; g3 < 3; ++g3) {
      v8f c8 = {};
      c8 = WMMA_F16(ah0, Whh1[0][g3], c8);
      c8 = WMMA_F16(ah1, Whh1[1][g3], c8);
      gh[g3] = c8;
    }

    float hnew1[8];
#pragma unroll
    for (int r = 0; r < 8; ++r) {
      int m = r + 8 * gg;
      float rr = fsig(gx[0][r] + vbih1[0] + gh[0][r] + vbhh1[0]);
      float zz = fsig(gx[1][r] + vbih1[1] + gh[1][r] + vbhh1[1]);
      float nn = ftanh(gx[2][r] + vbih1[2] + rr * (gh[2][r] + vbhh1[2]));
      float hp = (float)h1b[m * H_ + col];
      hnew1[r] = (1.0f - zz) * nn + zz * hp;
    }
    __syncthreads();                       // bar4: all h1 reads done
#pragma unroll
    for (int r = 0; r < 8; ++r) h1b[(r + 8 * gg) * H_ + col] = (_Float16)hnew1[r];
    // bar1 of next iteration publishes new h1
  }
  __syncthreads();

  // ================= head: hid = relu(h1 @ w1^T + b1) =================
  v16h ah0 = lds_a_frag(h1b, lane, 0);
  v16h ah1 = lds_a_frag(h1b, lane, 1);
  v16h W1f0 = *(const v16h*)(wsH + OFF_W1 + ((size_t)(0 * 4 + nj) * 32 + lane) * 16);
  v16h W1f1 = *(const v16h*)(wsH + OFF_W1 + ((size_t)(1 * 4 + nj) * 32 + lane) * 16);
  v8f hc = {};
  hc = WMMA_F16(ah0, W1f0, hc);
  hc = WMMA_F16(ah1, W1f1, hc);
  float b1v = b1[col];
#pragma unroll
  for (int r = 0; r < 8; ++r) {
    float v = hc[r] + b1v;
    h0b[(r + 8 * gg) * H_ + col] = (_Float16)(v > 0.0f ? v : 0.0f);  // reuse h0b
  }
  __syncthreads();

  // y = hid @ w2^T + b2  (1 output per batch row)
  if (tid < 16) {
    float s = b2[0];
#pragma unroll 8
    for (int k = 0; k < H_; ++k) s += (float)h0b[tid * H_ + k] * w2[k];
    out[bbase + tid] = s;
  }
}

extern "C" void kernel_launch(void* const* d_in, const int* in_sizes, int n_in,
                              void* d_out, int out_size, void* d_ws, size_t ws_size,
                              hipStream_t stream) {
  const float*         x    = (const float*)d_in[0];
  const unsigned char* xm   = (const unsigned char*)d_in[1];
  const float*         wih0 = (const float*)d_in[2];
  const float*         whh0 = (const float*)d_in[3];
  const float*         bih0 = (const float*)d_in[4];
  const float*         bhh0 = (const float*)d_in[5];
  const float*         wih1 = (const float*)d_in[6];
  const float*         whh1 = (const float*)d_in[7];
  const float*         bih1 = (const float*)d_in[8];
  const float*         bhh1 = (const float*)d_in[9];
  const float*         w1   = (const float*)d_in[10];
  const float*         b1   = (const float*)d_in[11];
  const float*         w2   = (const float*)d_in[12];
  const float*         b2   = (const float*)d_in[13];
  float*     out = (float*)d_out;
  _Float16*  wsH = (_Float16*)d_ws;

  // Pack weight B-fragments (f32 -> f16, WMMA 32x16 lane layout).
  gru_pack_bfrag<<<48, 256, 0, stream>>>(whh0, wsH + OFF_WHH0, 12);
  gru_pack_bfrag<<<48, 256, 0, stream>>>(wih1, wsH + OFF_WIH1, 12);
  gru_pack_bfrag<<<48, 256, 0, stream>>>(whh1, wsH + OFF_WHH1, 12);
  gru_pack_bfrag<<<16, 256, 0, stream>>>(w1,   wsH + OFF_W1,   4);

  // Fused 2-layer GRU + head: 128 blocks x 128 threads (16 batch rows/block).
  gru_fused<<<B_ / 16, 128, 0, stream>>>(x, xm, wih0, bih0, bhh0, bih1, bhh1,
                                         b1, w2, b2, wsH, out);
}
